// Attention_29858612641982
// MI455X (gfx1250) — matrix operand order repaired and checked by reading, not compiled, not run on hardware
//
#include <hip/hip_runtime.h>

#define DEV __device__ __forceinline__

// ---------------------------------------------------------------- types
typedef __attribute__((ext_vector_type(16))) __bf16 bf16x16;
typedef __attribute__((ext_vector_type(8)))  float  f32x8;
typedef __attribute__((ext_vector_type(4)))  unsigned int u32x4;

union Frag {
    bf16x16 v;
    u32x4   q[2];
};

constexpr int B_   = 16;
constexpr int C_   = 512;
constexpr int T_   = 1024;   // h*w
constexpr int NH_  = 8;
constexpr int EMB_ = 64;
constexpr int OC3_ = 3 * C_; // 1536

DEV unsigned short f2bf(float f) {
    union { float f; unsigned int u; } x; x.f = f;
    unsigned int r = x.u + 0x7FFFu + ((x.u >> 16) & 1u); // RNE
    return (unsigned short)(r >> 16);
}

DEV f32x8 wmma_bf16(bf16x16 a, bf16x16 b, f32x8 c) {
    // (neg_a, A, neg_b, B, c_mod, C, reuse_a, reuse_b)
    return __builtin_amdgcn_wmma_f32_16x16x32_bf16(false, a, false, b,
                                                   (short)0, c, false, false);
}

// A fragment: 16x32 bf16, row-major source, base at (m0,k0), leading dim lda.
// lane L: row = L&15, half h = L>>4; VGPR0-3 hold K = 8h+{0..7},
// VGPR4-7 hold K = 16+8h+{0..7}  -> two b128 loads.
DEV bf16x16 load_frag_a(const unsigned short* A, int lda, int lane) {
    const int r = lane & 15, h = lane >> 4;
    Frag f;
    f.q[0] = *(const u32x4*)(A + (size_t)r * lda + 8 * h);
    f.q[1] = *(const u32x4*)(A + (size_t)r * lda + 16 + 8 * h);
    return f.v;
}

// B fragment: 32x16 bf16 from N-major storage Bt[n][k] (ld = ldb), base (n0,k0).
// lane L: col = L&15, half h = L>>4; VGPR0-3 hold K = 8h+{0..7},
// VGPR4-7 hold K = 16+8h+{0..7}  -> two b128 loads.
DEV bf16x16 load_frag_b(const unsigned short* Bt, int ldb, int lane) {
    const int n = lane & 15, h = lane >> 4;
    const unsigned short* p = Bt + (size_t)n * ldb;
    Frag f;
    f.q[0] = *(const u32x4*)(p + 8 * h);
    f.q[1] = *(const u32x4*)(p + 16 + 8 * h);
    return f.v;
}

// ---------------------------------------------------------------- conversions
// x (b,c,t) f32  ->  xb (b,t,c) bf16   (B operand of QKV GEMM, N-major)
__global__ __launch_bounds__(256) void k_cvt_x(const float* __restrict__ x,
                                               unsigned short* __restrict__ xb) {
    size_t i = (size_t)blockIdx.x * blockDim.x + threadIdx.x;
    if (i >= (size_t)B_ * C_ * T_) return;
    int t = (int)(i & (T_ - 1));
    size_t bc = i >> 10;
    int c = (int)(bc & (C_ - 1));
    int b = (int)(bc >> 9);
    xb[((size_t)b * T_ + t) * C_ + c] = f2bf(x[i]);
}

// weights f32 -> bf16 (row-major copy)
__global__ __launch_bounds__(256) void k_cvt_w(const float* __restrict__ wq,
                                               const float* __restrict__ wp,
                                               unsigned short* __restrict__ wqb,
                                               unsigned short* __restrict__ wpb) {
    int i = blockIdx.x * blockDim.x + threadIdx.x;
    if (i < OC3_ * C_) wqb[i] = f2bf(wq[i]);
    if (i < C_ * C_)   wpb[i] = f2bf(wp[i]);
}

// ---------------------------------------------------------------- QKV GEMM
// qkv[o,t] = sum_c wqkv[o,c] * x[b,c,t] + bias[o]
// scatter: q,k -> (b,h,t,e) bf16 ; v -> (b,h,e,t) bf16 (transposed for PV GEMM)
__global__ __launch_bounds__(128) void k_qkv_gemm(
        const unsigned short* __restrict__ wqb,
        const unsigned short* __restrict__ xb,
        const float* __restrict__ bqkv,
        unsigned short* __restrict__ qw,
        unsigned short* __restrict__ kw,
        unsigned short* __restrict__ vw) {
    const int lane = threadIdx.x & 31;
    const int wid  = (int)((blockIdx.x * blockDim.x + threadIdx.x) >> 5);
    const int TPB  = (OC3_ / 32) * (T_ / 32);        // 48*32 tiles per batch
    const int b    = wid / TPB;
    const int rem  = wid % TPB;
    const int m0   = (rem / (T_ / 32)) * 32;
    const int n0   = (rem % (T_ / 32)) * 32;
    const unsigned short* Bb = xb + (size_t)b * T_ * C_;   // [T][C]

    f32x8 acc[2][2] = {};
    for (int k0 = 0; k0 < C_; k0 += 32) {
        __builtin_prefetch(Bb + (size_t)n0 * C_ + k0 + 32, 0, 1);
        __builtin_prefetch(wqb + (size_t)m0 * C_ + k0 + 32, 0, 1);
        bf16x16 a0 = load_frag_a(wqb + (size_t)m0 * C_ + k0, C_, lane);
        bf16x16 a1 = load_frag_a(wqb + (size_t)(m0 + 16) * C_ + k0, C_, lane);
        bf16x16 b0 = load_frag_b(Bb + (size_t)n0 * C_ + k0, C_, lane);
        bf16x16 b1 = load_frag_b(Bb + (size_t)(n0 + 16) * C_ + k0, C_, lane);
        acc[0][0] = wmma_bf16(a0, b0, acc[0][0]);
        acc[0][1] = wmma_bf16(a0, b1, acc[0][1]);
        acc[1][0] = wmma_bf16(a1, b0, acc[1][0]);
        acc[1][1] = wmma_bf16(a1, b1, acc[1][1]);
    }

    const int hf = lane >> 4, nn = lane & 15;
    for (int mi = 0; mi < 2; ++mi)
        for (int ni = 0; ni < 2; ++ni)
            for (int r = 0; r < 8; ++r) {
                int o = m0 + mi * 16 + r + 8 * hf;           // output channel
                int t = n0 + ni * 16 + nn;                   // spatial position
                unsigned short bv = f2bf(acc[mi][ni][r] + bqkv[o]);
                if (o < C_) {
                    int h = o >> 6, e = o & 63;
                    qw[(((size_t)b * NH_ + h) * T_ + t) * EMB_ + e] = bv;
                } else if (o < 2 * C_) {
                    int o2 = o - C_; int h = o2 >> 6, e = o2 & 63;
                    kw[(((size_t)b * NH_ + h) * T_ + t) * EMB_ + e] = bv;
                } else {
                    int o2 = o - 2 * C_; int h = o2 >> 6, e = o2 & 63;
                    vw[(((size_t)b * NH_ + h) * EMB_ + e) * T_ + t] = bv;
                }
            }
}

// ---------------------------------------------------------------- flash attention
// One wave = 16 query rows of one (b,head). Online softmax over 32-key blocks.
__global__ __launch_bounds__(128) void k_attn(
        const unsigned short* __restrict__ qws,
        const unsigned short* __restrict__ kws,
        const unsigned short* __restrict__ vws,
        unsigned short* __restrict__ outc) {
    __shared__ __align__(16) unsigned short Pt[4][16 * 32];  // P tile per wave

    const int lane = threadIdx.x & 31;
    const int wv   = threadIdx.x >> 5;
    const int wid  = (int)blockIdx.x * 4 + wv;
    const int b    = wid >> 9;          // 8 heads * 64 q-blocks = 512 waves/batch
    const int rem  = wid & 511;
    const int h    = rem >> 6;
    const int q0   = (rem & 63) << 4;
    const size_t bh = (size_t)b * NH_ + h;

    const unsigned short* qb = qws + (bh * T_ + q0) * EMB_;  // (t,e)
    const unsigned short* kb = kws + bh * T_ * EMB_;         // (t,e)
    const unsigned short* vb = vws + bh * EMB_ * T_;         // (e,t)

    const int row = lane & 15, hf = lane >> 4;
    const float scale = 0.125f;                              // 1/sqrt(64)

    // q cached in registers: two A fragments (e = 0..31, 32..63)
    bf16x16 aq0 = load_frag_a(qb + 0, EMB_, lane);
    bf16x16 aq1 = load_frag_a(qb + 32, EMB_, lane);

    f32x8 oacc[4] = {};
    float mr[8], lr[8];
    for (int r = 0; r < 8; ++r) { mr[r] = -1e30f; lr[r] = 0.f; }

    for (int key0 = 0; key0 < T_; key0 += 32) {
        // S = q * k^T  (two 16x16 tiles covering 32 keys; K = emb = 64)
        f32x8 s0 = {}, s1 = {};
        s0 = wmma_bf16(aq0, load_frag_b(kb + (size_t)key0 * EMB_ + 0,  EMB_, lane), s0);
        s0 = wmma_bf16(aq1, load_frag_b(kb + (size_t)key0 * EMB_ + 32, EMB_, lane), s0);
        s1 = wmma_bf16(aq0, load_frag_b(kb + (size_t)(key0 + 16) * EMB_ + 0,  EMB_, lane), s1);
        s1 = wmma_bf16(aq1, load_frag_b(kb + (size_t)(key0 + 16) * EMB_ + 32, EMB_, lane), s1);

        float p0[8], p1[8], alpha[8];
        for (int r = 0; r < 8; ++r) {
            float a = s0[r] * scale, c = s1[r] * scale;
            // block row-max across 32 keys: per-lane max, then 16-lane tree
            float t = fmaxf(a, c);
            t = fmaxf(t, __shfl_xor(t, 1, 16));
            t = fmaxf(t, __shfl_xor(t, 2, 16));
            t = fmaxf(t, __shfl_xor(t, 4, 16));
            t = fmaxf(t, __shfl_xor(t, 8, 16));
            float mn = fmaxf(mr[r], t);
            alpha[r] = __expf(mr[r] - mn);
            mr[r] = mn;
            p0[r] = __expf(a - mn);
            p1[r] = __expf(c - mn);
            float rs = p0[r] + p1[r];
            rs += __shfl_xor(rs, 1, 16);
            rs += __shfl_xor(rs, 2, 16);
            rs += __shfl_xor(rs, 4, 16);
            rs += __shfl_xor(rs, 8, 16);
            lr[r] = lr[r] * alpha[r] + rs;
        }
        for (int j = 0; j < 4; ++j)
            for (int r = 0; r < 8; ++r)
                oacc[j][r] *= alpha[r];

        // C-layout -> A-layout transpose of P through LDS (16x32 bf16 tile)
        for (int r = 0; r < 8; ++r) {
            int m = r + 8 * hf;
            Pt[wv][m * 32 + row]      = f2bf(p0[r]);
            Pt[wv][m * 32 + 16 + row] = f2bf(p1[r]);
        }
        // wave-local: drain DS stores, fence scheduler; same-wave LDS ops are
        // in-order so next iteration's stores cannot pass this block's loads.
        asm volatile("s_wait_dscnt 0" ::: "memory");
        __builtin_amdgcn_wave_barrier();

        Frag fp;
        fp.q[0] = *(const u32x4*)&Pt[wv][row * 32 + 8 * hf];
        fp.q[1] = *(const u32x4*)&Pt[wv][row * 32 + 16 + 8 * hf];

        // O += P * V  (v stored (e,t): contiguous keys per lane)
        for (int j = 0; j < 4; ++j) {
            bf16x16 bvf = load_frag_b(vb + (size_t)(j * 16) * T_ + key0, T_, lane);
            oacc[j] = wmma_bf16(fp.v, bvf, oacc[j]);
        }
    }

    // normalize and emit as (b, t, c) bf16 for the proj GEMM (N-major B operand)
    for (int r = 0; r < 8; ++r) {
        float inv = 1.0f / lr[r];
        for (int j = 0; j < 4; ++j) oacc[j][r] *= inv;
    }
    for (int j = 0; j < 4; ++j)
        for (int r = 0; r < 8; ++r) {
            int m = r + 8 * hf;
            int cch = h * EMB_ + j * 16 + row;
            outc[((size_t)b * T_ + q0 + m) * C_ + cch] = f2bf(oacc[j][r]);
        }
}

// ---------------------------------------------------------------- proj GEMM
// y[b,o,t] = sum_c wproj[o,c] * outc[b,t,c] + bproj[o]   (f32 output)
__global__ __launch_bounds__(128) void k_proj_gemm(
        const unsigned short* __restrict__ wpb,
        const unsigned short* __restrict__ outc,
        const float* __restrict__ bproj,
        float* __restrict__ y) {
    const int lane = threadIdx.x & 31;
    const int wid  = (int)((blockIdx.x * blockDim.x + threadIdx.x) >> 5);
    const int TPB  = (C_ / 32) * (T_ / 32);          // 16*32 tiles per batch
    const int b    = wid / TPB;
    const int rem  = wid % TPB;
    const int m0   = (rem / (T_ / 32)) * 32;
    const int n0   = (rem % (T_ / 32)) * 32;
    const unsigned short* Bb = outc + (size_t)b * T_ * C_;  // [T][C]

    f32x8 acc[2][2] = {};
    for (int k0 = 0; k0 < C_; k0 += 32) {
        __builtin_prefetch(Bb + (size_t)n0 * C_ + k0 + 32, 0, 1);
        bf16x16 a0 = load_frag_a(wpb + (size_t)m0 * C_ + k0, C_, lane);
        bf16x16 a1 = load_frag_a(wpb + (size_t)(m0 + 16) * C_ + k0, C_, lane);
        bf16x16 b0 = load_frag_b(Bb + (size_t)n0 * C_ + k0, C_, lane);
        bf16x16 b1 = load_frag_b(Bb + (size_t)(n0 + 16) * C_ + k0, C_, lane);
        acc[0][0] = wmma_bf16(a0, b0, acc[0][0]);
        acc[0][1] = wmma_bf16(a0, b1, acc[0][1]);
        acc[1][0] = wmma_bf16(a1, b0, acc[1][0]);
        acc[1][1] = wmma_bf16(a1, b1, acc[1][1]);
    }

    const int hf = lane >> 4, nn = lane & 15;
    for (int mi = 0; mi < 2; ++mi)
        for (int ni = 0; ni < 2; ++ni)
            for (int r = 0; r < 8; ++r) {
                int o = m0 + mi * 16 + r + 8 * hf;
                int t = n0 + ni * 16 + nn;
                y[((size_t)b * C_ + o) * T_ + t] = acc[mi][ni][r] + bproj[o];
            }
}

// ---------------------------------------------------------------- launch
extern "C" void kernel_launch(void* const* d_in, const int* in_sizes, int n_in,
                              void* d_out, int out_size, void* d_ws, size_t ws_size,
                              hipStream_t stream) {
    const float* x      = (const float*)d_in[0];
    const float* w_qkv  = (const float*)d_in[1];
    const float* b_qkv  = (const float*)d_in[2];
    const float* w_proj = (const float*)d_in[3];
    const float* b_proj = (const float*)d_in[4];
    float* out = (float*)d_out;

    char* ws = (char*)d_ws;
    size_t off = 0;
    auto alloc = [&](size_t bytes) -> void* {
        void* p = ws + off;
        off += (bytes + 255) & ~(size_t)255;
        return p;
    };
    unsigned short* xb    = (unsigned short*)alloc((size_t)B_ * T_ * C_ * 2);
    unsigned short* wqkvb = (unsigned short*)alloc((size_t)OC3_ * C_ * 2);
    unsigned short* wprojb= (unsigned short*)alloc((size_t)C_ * C_ * 2);
    unsigned short* qw    = (unsigned short*)alloc((size_t)B_ * NH_ * T_ * EMB_ * 2);
    unsigned short* kw    = (unsigned short*)alloc((size_t)B_ * NH_ * T_ * EMB_ * 2);
    unsigned short* vw    = (unsigned short*)alloc((size_t)B_ * NH_ * EMB_ * T_ * 2);
    unsigned short* outc  = (unsigned short*)alloc((size_t)B_ * T_ * C_ * 2);

    // 1) fp32 -> bf16 conversions (x is transposed to (b,t,c))
    {
        size_t n = (size_t)B_ * C_ * T_;
        k_cvt_x<<<dim3((unsigned)((n + 255) / 256)), dim3(256), 0, stream>>>(x, xb);
        int nw = OC3_ * C_;
        k_cvt_w<<<dim3((nw + 255) / 256), dim3(256), 0, stream>>>(w_qkv, w_proj, wqkvb, wprojb);
    }
    // 2) QKV GEMM + scatter to per-head layouts
    {
        int waves = B_ * (OC3_ / 32) * (T_ / 32);   // 24576
        k_qkv_gemm<<<dim3(waves / 4), dim3(128), 0, stream>>>(wqkvb, xb, b_qkv, qw, kw, vw);
    }
    // 3) flash attention
    {
        int waves = B_ * NH_ * (T_ / 16);           // 8192
        k_attn<<<dim3(waves / 4), dim3(128), 0, stream>>>(qw, kw, vw, outc);
    }
    // 4) output projection
    {
        int waves = B_ * (C_ / 32) * (T_ / 32);     // 8192
        k_proj_gemm<<<dim3(waves / 4), dim3(128), 0, stream>>>(wprojb, outc, b_proj, out);
    }
    (void)in_sizes; (void)n_in; (void)out_size; (void)ws_size;
}
